// Quantizer_738734375640
// MI455X (gfx1250) — matrix operand order repaired and checked by reading, not compiled, not run on hardware
//
#include <hip/hip_runtime.h>
#include <hip/hip_bf16.h>

typedef __attribute__((ext_vector_type(16))) _Float16 v16h;
typedef __attribute__((ext_vector_type(8)))  _Float16 v8h;
typedef __attribute__((ext_vector_type(8)))  float    v8f;

// Problem sizes
#define N_PIX   32768      // 32 * 32 * 32 pixels
#define C_DIM   256
#define K_CB    1024
#define QUANT_ELEMS 8388608 // 32*256*32*32
#define LOSS_OFF    8388608
#define IDX_OFF     8388609

// workspace layout (bytes); ws assumed >= ~18 MB
#define WS_XF16   0                       // 32768*256*2  = 16777216
#define WS_CB16   16777216                // 1024*256*2   = 524288
#define WS_E2     (16777216 + 524288)     // 1024*4       = 4096
#define WS_IDX    (16777216 + 524288 + 4096) // 32768*4   = 131072

__global__ void zero_loss_kernel(float* __restrict__ out) {
  out[LOSS_OFF] = 0.0f;
}

// x [32][256][1024] f32  ->  xf16 [n=b*1024+p][c] f16 (row-contiguous for WMMA A)
__global__ void transpose_x_kernel(const float* __restrict__ x,
                                   _Float16* __restrict__ xf16) {
  __shared__ float tile[32][33];
  int b  = blockIdx.y;
  int tp = (blockIdx.x & 31) << 5;   // pixel tile base
  int tc = (blockIdx.x >> 5) << 5;   // channel tile base
  int tx = threadIdx.x;              // 0..31
  int ty = threadIdx.y;              // 0..7
  const float* src = x + ((size_t)(((b << 8) + tc)) << 10) + tp;
#pragma unroll
  for (int r = 0; r < 4; ++r) {
    int cl = ty + (r << 3);
    tile[cl][tx] = src[((size_t)cl << 10) + tx];   // coalesced in p
  }
  __syncthreads();
#pragma unroll
  for (int r = 0; r < 4; ++r) {
    int pl = ty + (r << 3);
    int n  = (b << 10) + tp + pl;
    xf16[((size_t)n << 8) + tc + tx] = (_Float16)tile[tx][pl]; // contiguous in c
  }
}

// codebook f32 -> f16 copy + e2[k] = sum_c cb[k][c]^2 (exact f32)
__global__ void prep_codebook_kernel(const float* __restrict__ cb,
                                     _Float16* __restrict__ cb16,
                                     float* __restrict__ e2) {
  __shared__ float red[256];
  int k = blockIdx.x, c = threadIdx.x;
  float v = cb[(k << 8) + c];
  cb16[(k << 8) + c] = (_Float16)v;
  red[c] = v * v;
  __syncthreads();
  for (int s = 128; s > 0; s >>= 1) {
    if (c < s) red[c] += red[c + s];
    __syncthreads();
  }
  if (c == 0) e2[k] = red[0];
}

// One wave owns 16 pixel rows; streams all 64 codebook 16-column tiles,
// processed two tiles per iteration with independent accumulator chains so
// the scheduler can overlap B-tile loads of one chain with WMMAs of the other.
// dist(n,k) argmin is invariant to |x|^2, so track e2[k] - 2*(x . c_k).
__global__ void __launch_bounds__(128)
argmin_kernel(const _Float16* __restrict__ xf16,
              const _Float16* __restrict__ cb16,
              const float* __restrict__ e2,
              int* __restrict__ idx) {
  int lane = threadIdx.x & 31;
  int wave = (blockIdx.x * blockDim.x + threadIdx.x) >> 5;
  int row0 = wave << 4;            // 16 rows per wave
  int m    = lane & 15;
  int hi   = lane >> 4;            // half-wave select

  // A matrix (16x32 f16 per k-step), ISA layout:
  // lanes 0-15: row m, K 0..7 then 16..23 ; lanes 16-31: row m, K 8..15 then 24..31
  v16h a[8];
  const _Float16* arow = xf16 + ((size_t)(row0 + m) << 8);
#pragma unroll
  for (int ch = 0; ch < 8; ++ch) {
    const _Float16* p = arow + (ch << 5) + (hi << 3);
    v8h lo  = *(const v8h*)(p);
    v8h hih = *(const v8h*)(p + 16);
    a[ch] = __builtin_shufflevector(lo, hih,
              0,1,2,3,4,5,6,7,8,9,10,11,12,13,14,15);
  }

  float minv[8];
  int   mini[8];
#pragma unroll
  for (int i = 0; i < 8; ++i) { minv[i] = 3.0e38f; mini[i] = 0; }

  for (int kt = 0; kt < 32; ++kt) {
    int col0 = (kt << 5) + m;        // lower column tile (checked first: tie-break)
    int col1 = col0 + 16;            // upper column tile
    const _Float16* brow0 = cb16 + ((size_t)col0 << 8) + (hi << 3); // K 0..7,16..23 / 8..15,24..31
    const _Float16* brow1 = cb16 + ((size_t)col1 << 8) + (hi << 3);
    v8f acc0 = {};
    v8f acc1 = {};
#pragma unroll
    for (int ch = 0; ch < 8; ++ch) {
      v16h b0 = __builtin_shufflevector(*(const v8h*)(brow0 + (ch << 5)), *(const v8h*)(brow0 + (ch << 5) + 16), 0,1,2,3,4,5,6,7,8,9,10,11,12,13,14,15);   // two 16B chunks
      v16h b1 = __builtin_shufflevector(*(const v8h*)(brow1 + (ch << 5)), *(const v8h*)(brow1 + (ch << 5) + 16), 0,1,2,3,4,5,6,7,8,9,10,11,12,13,14,15);
      acc0 = __builtin_amdgcn_wmma_f32_16x16x32_f16(
               false, a[ch], false, b0, (short)0, acc0, false, false);
      acc1 = __builtin_amdgcn_wmma_f32_16x16x32_f16(
               false, a[ch], false, b1, (short)0, acc1, false, false);
    }
    float e2v0 = e2[col0];
    float e2v1 = e2[col1];
#pragma unroll
    for (int i = 0; i < 8; ++i) {
      float d0 = e2v0 - 2.0f * acc0[i];
      if (d0 < minv[i]) { minv[i] = d0; mini[i] = col0; }
      float d1 = e2v1 - 2.0f * acc1[i];
      if (d1 < minv[i]) { minv[i] = d1; mini[i] = col1; }
    }
  }

  // C/D layout: acc[i] -> row (row0 + i + 8*hi), col striped over 16 lanes of the half.
  // Reduce min over the 16 lanes of each half (xor masks 1..8 stay in-half).
#pragma unroll
  for (int i = 0; i < 8; ++i) {
    float v = minv[i]; int id = mini[i];
#pragma unroll
    for (int off = 1; off < 16; off <<= 1) {
      float ov = __shfl_xor(v, off, 32);
      int   oi = __shfl_xor(id, off, 32);
      if (ov < v || (ov == v && oi < id)) { v = ov; id = oi; } // first-index tie-break
    }
    if (m == i) idx[row0 + i + (hi << 3)] = id;
  }
}

// quant_out gather (exact f32) + fused loss = 2 * mean((c_idx - x)^2)
__global__ void quant_loss_kernel(const float* __restrict__ x,
                                  const float* __restrict__ cb,
                                  const int* __restrict__ idx,
                                  float* __restrict__ out) {
  __shared__ float red[256];
  int o  = blockIdx.x * 256 + threadIdx.x;   // over [b][c][p], coalesced in p
  int p  = o & 1023;
  int bc = o >> 10;
  int c  = bc & 255;
  int b  = bc >> 8;
  int id = idx[(b << 10) + p];
  float q  = cb[(id << 8) + c];              // gather from L2-resident codebook
  float xv = x[o];
  out[o] = q;
  float d = q - xv;
  red[threadIdx.x] = d * d;
  __syncthreads();
  for (int s = 128; s > 0; s >>= 1) {
    if ((int)threadIdx.x < s) red[threadIdx.x] += red[threadIdx.x + s];
    __syncthreads();
  }
  if (threadIdx.x == 0)
    atomicAdd(out + LOSS_OFF, red[0] * (2.0f / 8388608.0f));
}

__global__ void write_idx_kernel(const int* __restrict__ idx,
                                 float* __restrict__ out) {
  int n = blockIdx.x * 256 + threadIdx.x;
  out[IDX_OFF + n] = (float)idx[n];
}

extern "C" void kernel_launch(void* const* d_in, const int* in_sizes, int n_in,
                              void* d_out, int out_size, void* d_ws, size_t ws_size,
                              hipStream_t stream) {
  const float* x  = (const float*)d_in[0];   // [32,256,32,32]
  const float* cb = (const float*)d_in[1];   // [1024,256]
  float* out = (float*)d_out;
  char*  ws  = (char*)d_ws;
  _Float16* xf16 = (_Float16*)(ws + WS_XF16);
  _Float16* cb16 = (_Float16*)(ws + WS_CB16);
  float*    e2   = (float*)(ws + WS_E2);
  int*      idx  = (int*)(ws + WS_IDX);

  zero_loss_kernel<<<1, 1, 0, stream>>>(out);
  transpose_x_kernel<<<dim3(256, 32), dim3(32, 8), 0, stream>>>(x, xf16);
  prep_codebook_kernel<<<K_CB, 256, 0, stream>>>(cb, cb16, e2);
  argmin_kernel<<<N_PIX / 64, 128, 0, stream>>>(xf16, cb16, e2, idx);
  quant_loss_kernel<<<QUANT_ELEMS / 256, 256, 0, stream>>>(x, cb, idx, out);
  write_idx_kernel<<<N_PIX / 256, 256, 0, stream>>>(idx, out);
}
